// Qwen3Attention_67680094651083
// MI455X (gfx1250) — compile-verified, matrix operand order checked by reading, not run on hardware
//
#include <hip/hip_runtime.h>

typedef unsigned short u16;
typedef __attribute__((ext_vector_type(16))) __bf16 v16bf;
typedef __attribute__((ext_vector_type(8)))  float  v8f;

#define B_   2
#define S_   2048
#define H_   2048
#define NH_  16
#define NKV_ 8
#define D_   128
#define WIN_ 1024

// round-to-nearest-even f32 -> bf16 bits
__device__ __forceinline__ u16 f2bf(float x) {
  unsigned int u = __float_as_uint(x);
  u += 0x7fffu + ((u >> 16) & 1u);
  return (u16)(u >> 16);
}

__device__ __forceinline__ v8f wmma_bf16(v16bf a, v16bf b, v8f c) {
  return __builtin_amdgcn_wmma_f32_16x16x32_bf16(false, a, false, b, (short)0, c,
                                                 false, false);
}

// gfx1250 async global->LDS copy, 16B per lane, tracked by ASYNCcnt.
// LDS flat address low 32 bits == LDS byte offset (aperture mapping).
__device__ __forceinline__ void async_ld_b128(unsigned lds_off, const void* g) {
  asm volatile("global_load_async_to_lds_b128 %0, %1, off"
               :: "v"(lds_off), "v"(g) : "memory");
}

// 16x32 bf16 A/B fragment, outer index (M for A, N for B) = lane&15 with
// stride `ld`, K contiguous in memory (ISA 7.12.2 layout).
__device__ __forceinline__ v16bf frag_ld_rowmajor(const __bf16* base, int ld, int lane) {
  const __bf16* p = base + (size_t)(lane & 15) * ld + (((lane >> 4) & 1) << 3);
  v16bf f;
#pragma unroll
  for (int i = 0; i < 8; ++i) {
    int k = 2 * i + ((i >= 4) ? 8 : 0);
    f[2 * i]     = p[k];
    f[2 * i + 1] = p[k + 1];
  }
  return f;
}

// ---------------------------------------------------------------- convert
__global__ void k_f32_to_bf16(const float* __restrict__ in, u16* __restrict__ out,
                              long n) {
  long i = (long)blockIdx.x * blockDim.x + threadIdx.x;
  if (i < n) out[i] = f2bf(in[i]);
}

// ---------------------------------------------------------------- GEMM
// C[M,N](f32) = A[M,K](bf16, row-major) * B[N,K](bf16, row-major)^T
// block = (32,8). Block tile: 128(M) x 64(N).  K staged 32-wide through LDS
// with double-buffered async-to-LDS copies (3 async instrs / wave / tile).
__global__ void k_gemm_bf16(const u16* __restrict__ Au, const u16* __restrict__ Bu,
                            float* __restrict__ C, int M, int N, int K) {
  __shared__ u16 Atile[2][128 * 32];
  __shared__ u16 Btile[2][64 * 32];
  int lane = threadIdx.x, wave = threadIdx.y;
  int tid = wave * 32 + lane;                  // 0..255
  int m0 = blockIdx.y * 128;
  int n0 = blockIdx.x * 64;
  const u16* Ag = Au + (size_t)m0 * K;
  const u16* Bg = Bu + (size_t)n0 * K;

  int row = tid >> 2, ch = (tid & 3) * 8;      // 4 x 16B chunks per 32-elem row

  v8f z = {0.f, 0.f, 0.f, 0.f, 0.f, 0.f, 0.f, 0.f};
  v8f acc[4];
#pragma unroll
  for (int j = 0; j < 4; ++j) acc[j] = z;

  int nk = K >> 5;
  // prologue: tile 0 into buffer 0
  {
    async_ld_b128((unsigned)(uintptr_t)&Atile[0][row * 32 + ch],
                  Ag + (size_t)row * K + ch);
    async_ld_b128((unsigned)(uintptr_t)&Atile[0][(row + 64) * 32 + ch],
                  Ag + (size_t)(row + 64) * K + ch);
    async_ld_b128((unsigned)(uintptr_t)&Btile[0][row * 32 + ch],
                  Bg + (size_t)row * K + ch);
  }

  for (int kt = 0; kt < nk; ++kt) {
    int buf = kt & 1;
    if (kt + 1 < nk) {
      size_t koff = (size_t)(kt + 1) << 5;
      int nb = buf ^ 1;
      async_ld_b128((unsigned)(uintptr_t)&Atile[nb][row * 32 + ch],
                    Ag + (size_t)row * K + koff + ch);
      async_ld_b128((unsigned)(uintptr_t)&Atile[nb][(row + 64) * 32 + ch],
                    Ag + (size_t)(row + 64) * K + koff + ch);
      async_ld_b128((unsigned)(uintptr_t)&Btile[nb][row * 32 + ch],
                    Bg + (size_t)row * K + koff + ch);
      asm volatile("s_wait_asynccnt 3" ::: "memory");   // old batch done, new in flight
    } else {
      asm volatile("s_wait_asynccnt 0" ::: "memory");
    }
    __syncthreads();

    // batch all fragment loads (one LDS clause, one wait), then 4 WMMAs back-to-back
    v16bf af = frag_ld_rowmajor((const __bf16*)&Atile[buf][wave * 16 * 32], 32, lane);
    v16bf bf0 = frag_ld_rowmajor((const __bf16*)&Btile[buf][0 * 16 * 32], 32, lane);
    v16bf bf1 = frag_ld_rowmajor((const __bf16*)&Btile[buf][1 * 16 * 32], 32, lane);
    v16bf bf2 = frag_ld_rowmajor((const __bf16*)&Btile[buf][2 * 16 * 32], 32, lane);
    v16bf bf3 = frag_ld_rowmajor((const __bf16*)&Btile[buf][3 * 16 * 32], 32, lane);
    acc[0] = wmma_bf16(af, bf0, acc[0]);
    acc[1] = wmma_bf16(af, bf1, acc[1]);
    acc[2] = wmma_bf16(af, bf2, acc[2]);
    acc[3] = wmma_bf16(af, bf3, acc[3]);
    __syncthreads();                                    // safe to overwrite buf at kt+2
  }

  int hl = (lane >> 4) & 1, ln = lane & 15;
  int mw = m0 + wave * 16;
#pragma unroll
  for (int j = 0; j < 4; ++j)
#pragma unroll
    for (int r = 0; r < 8; ++r)
      C[(size_t)(mw + r + 8 * hl) * N + n0 + j * 16 + ln] = acc[j][r];
}

// ---------------------------------------------------------------- norm+rope
// X: (B*S, nheads*128) f32 -> bf16, scaled.
// vtrans==0: out (B, nheads, S, 128);  vtrans==1: out (B, nheads, 128, S)
__global__ void k_normrope(const float* __restrict__ X, const float* __restrict__ cosb,
                           const float* __restrict__ sinb, const float* __restrict__ w,
                           u16* __restrict__ out, int nheads, int do_norm, int do_rope,
                           int vtrans, float scale) {
  int lane = threadIdx.x;                 // 32 lanes, 4 elems each
  int idx = blockIdx.x;                   // (b*S+s)*nheads + h
  int h = idx % nheads;
  int bs = idx / nheads;
  const float* x = X + (size_t)bs * (nheads * D_) + h * D_;
  int d = lane * 4;

  float x0 = x[d + 0], x1 = x[d + 1], x2 = x[d + 2], x3 = x[d + 3];
  float ss = x0 * x0 + x1 * x1 + x2 * x2 + x3 * x3;
#pragma unroll
  for (int off = 1; off < 32; off <<= 1) ss += __shfl_xor(ss, off, 32);
  float rstd = do_norm ? rsqrtf(ss * (1.0f / D_) + 1e-6f) : 1.0f;

  float w0 = 1.f, w1 = 1.f, w2 = 1.f, w3 = 1.f;
  if (do_norm) { w0 = w[d]; w1 = w[d + 1]; w2 = w[d + 2]; w3 = w[d + 3]; }
  float n0 = x0 * rstd * w0, n1 = x1 * rstd * w1, n2 = x2 * rstd * w2, n3 = x3 * rstd * w3;

  float r0 = n0, r1 = n1, r2 = n2, r3 = n3;
  if (do_rope) {
    int pd = (d + 64) & (D_ - 1);
    float p0 = x[pd], p1 = x[pd + 1], p2 = x[pd + 2], p3 = x[pd + 3];
    float q0 = 1.f, q1 = 1.f, q2 = 1.f, q3 = 1.f;
    if (do_norm) { q0 = w[pd]; q1 = w[pd + 1]; q2 = w[pd + 2]; q3 = w[pd + 3]; }
    float sgn = (lane < 16) ? -1.0f : 1.0f;
    const float* cp = cosb + (size_t)bs * D_ + d;
    const float* sp = sinb + (size_t)bs * D_ + d;
    r0 = n0 * cp[0] + sgn * (p0 * rstd * q0) * sp[0];
    r1 = n1 * cp[1] + sgn * (p1 * rstd * q1) * sp[1];
    r2 = n2 * cp[2] + sgn * (p2 * rstd * q2) * sp[2];
    r3 = n3 * cp[3] + sgn * (p3 * rstd * q3) * sp[3];
  }
  int b = bs / S_, s = bs % S_;
  if (vtrans) {
    size_t o = ((size_t)(b * nheads + h) * D_ + d) * S_ + s;
    out[o + 0 * S_] = f2bf(r0 * scale);
    out[o + 1 * S_] = f2bf(r1 * scale);
    out[o + 2 * S_] = f2bf(r2 * scale);
    out[o + 3 * S_] = f2bf(r3 * scale);
  } else {
    size_t o = ((size_t)(b * nheads + h) * S_ + s) * D_ + d;
    out[o + 0] = f2bf(r0 * scale);
    out[o + 1] = f2bf(r1 * scale);
    out[o + 2] = f2bf(r2 * scale);
    out[o + 3] = f2bf(r3 * scale);
  }
}

// ---------------------------------------------------------------- attention
// Qb/Kb: (B, nh|nkv, S, 128) bf16, Q pre-scaled by d^-0.5 * log2(e).
// Vt: (B, nkv, 128, S) bf16 (d-major) -> P*V uses contiguous-K fragments.
// block = (32,4): 4 independent waves, each owns a 16-query tile.
__global__ void k_attn(const u16* __restrict__ Qu, const u16* __restrict__ Ku,
                       const u16* __restrict__ Vu, u16* __restrict__ Attn) {
  __shared__ u16 plds_all[4][16 * 32];
  int lane = threadIdx.x;
  int wave = threadIdx.y;
  u16* plds = plds_all[wave];

  int bh = blockIdx.y;
  int b = bh >> 4, h = bh & 15;
  int q0 = (blockIdx.x * 4 + wave) * 16;

  const __bf16* Q = (const __bf16*)Qu + ((size_t)(b * NH_ + h) * S_ + q0) * D_;
  const __bf16* K = (const __bf16*)Ku + ((size_t)(b * NKV_ + (h >> 1)) * S_) * D_;
  const __bf16* V = (const __bf16*)Vu + ((size_t)(b * NKV_ + (h >> 1)) * D_) * S_;

  v16bf qf[4];
#pragma unroll
  for (int c = 0; c < 4; ++c) qf[c] = frag_ld_rowmajor(Q + c * 32, D_, lane);

  v8f z = {0.f, 0.f, 0.f, 0.f, 0.f, 0.f, 0.f, 0.f};
  v8f acc[8];
  float m[8], l[8];
#pragma unroll
  for (int f = 0; f < 8; ++f) acc[f] = z;
#pragma unroll
  for (int r = 0; r < 8; ++r) { m[r] = -1e30f; l[r] = 0.0f; }

  int hl = (lane >> 4) & 1, ln = lane & 15;
  int kb0 = (q0 > (WIN_ - 1)) ? ((q0 - (WIN_ - 1)) & ~31) : 0;

  for (int kb = kb0; kb < q0 + 16; kb += 32) {
    // S tile: 16 queries x 32 keys, Kdim = 128.
    // Load all 8 K fragments first, then issue the 8 WMMAs back-to-back.
    v16bf kf[8];
#pragma unroll
    for (int c = 0; c < 4; ++c) {
      kf[2 * c]     = frag_ld_rowmajor(K + (size_t)kb * D_ + c * 32, D_, lane);
      kf[2 * c + 1] = frag_ld_rowmajor(K + (size_t)(kb + 16) * D_ + c * 32, D_, lane);
    }
    v8f s0 = z, s1 = z;
#pragma unroll
    for (int c = 0; c < 4; ++c) {
      s0 = wmma_bf16(qf[c], kf[2 * c], s0);
      s1 = wmma_bf16(qf[c], kf[2 * c + 1], s1);
    }

    int col0 = kb + ln;
    int col1 = col0 + 16;
    int rbase = q0 + 8 * hl;
#pragma unroll
    for (int r = 0; r < 8; ++r) {
      int row = rbase + r;
      bool v0 = (col0 <= row) && (row - col0 < WIN_);
      bool v1 = (col1 <= row) && (row - col1 < WIN_);
      float e0 = v0 ? s0[r] : -1e30f;
      float e1 = v1 ? s1[r] : -1e30f;
      float mx = fmaxf(e0, e1);
#pragma unroll
      for (int off = 1; off < 16; off <<= 1) mx = fmaxf(mx, __shfl_xor(mx, off, 32));
      float nm = fmaxf(m[r], mx);
      float corr = exp2f(m[r] - nm);               // scores already in log2e domain
      float p0 = v0 ? exp2f(s0[r] - nm) : 0.0f;
      float p1 = v1 ? exp2f(s1[r] - nm) : 0.0f;
      float ps = p0 + p1;
#pragma unroll
      for (int off = 1; off < 16; off <<= 1) ps += __shfl_xor(ps, off, 32);
      l[r] = l[r] * corr + ps;
      m[r] = nm;
#pragma unroll
      for (int f = 0; f < 8; ++f) acc[f][r] *= corr;
      // C-layout -> LDS, row-major 16x32 bf16 (re-read as A fragment)
      int lr = r + 8 * hl;
      plds[lr * 32 + (col0 - kb)] = f2bf(p0);
      plds[lr * 32 + (col1 - kb)] = f2bf(p1);
    }
    asm volatile("s_wait_dscnt 0" ::: "memory");

    // Load P + all 8 V^T fragments, then 8 accumulation WMMAs back-to-back.
    v16bf pf = frag_ld_rowmajor((const __bf16*)plds, 32, lane);
    v16bf vf[8];
#pragma unroll
    for (int f = 0; f < 8; ++f)
      vf[f] = frag_ld_rowmajor(V + (size_t)(f * 16) * S_ + kb, S_, lane);
#pragma unroll
    for (int f = 0; f < 8; ++f)
      acc[f] = wmma_bf16(pf, vf[f], acc[f]);
  }

  // epilogue: normalize and write (b, s, h*128+d) bf16
#pragma unroll
  for (int r = 0; r < 8; ++r) {
    int row = q0 + 8 * hl + r;
    float inv = 1.0f / l[r];
#pragma unroll
    for (int f = 0; f < 8; ++f) {
      size_t o = (size_t)(b * S_ + row) * (NH_ * D_) + h * D_ + f * 16 + ln;
      Attn[o] = f2bf(acc[f][r] * inv);
    }
  }
}

// ---------------------------------------------------------------- launch
extern "C" void kernel_launch(void* const* d_in, const int* in_sizes, int n_in,
                              void* d_out, int out_size, void* d_ws, size_t ws_size,
                              hipStream_t stream) {
  (void)in_sizes; (void)n_in; (void)out_size; (void)ws_size;
  const float* hs   = (const float*)d_in[0];
  const float* cosb = (const float*)d_in[1];
  const float* sinb = (const float*)d_in[2];
  const float* Wq = (const float*)d_in[3];
  const float* Wk = (const float*)d_in[4];
  const float* Wv = (const float*)d_in[5];
  const float* Wo = (const float*)d_in[6];
  const float* qnw = (const float*)d_in[7];
  const float* knw = (const float*)d_in[8];
  float* out = (float*)d_out;

  char* ws = (char*)d_ws;
  size_t off = 0;
  auto alloc = [&](size_t bytes) { void* p = ws + off; off += (bytes + 255) & ~(size_t)255; return p; };

  const long nX = (long)B_ * S_ * H_;           // 8.4M
  const long nWq = (long)NH_ * D_ * H_;         // 4.2M
  const long nWkv = (long)NKV_ * D_ * H_;       // 2.1M
  u16* Xb  = (u16*)alloc(nX * 2);
  u16* Wqb = (u16*)alloc(nWq * 2);
  u16* Wkb = (u16*)alloc(nWkv * 2);
  u16* Wvb = (u16*)alloc(nWkv * 2);
  u16* Wob = (u16*)alloc(nWq * 2);
  float* Qf = (float*)alloc((size_t)B_ * S_ * NH_ * D_ * 4);
  float* Kf = (float*)alloc((size_t)B_ * S_ * NKV_ * D_ * 4);
  float* Vf = (float*)alloc((size_t)B_ * S_ * NKV_ * D_ * 4);
  u16* Qb = (u16*)alloc((size_t)B_ * NH_ * S_ * D_ * 2);
  u16* Kb = (u16*)alloc((size_t)B_ * NKV_ * S_ * D_ * 2);
  u16* Vt = (u16*)alloc((size_t)B_ * NKV_ * D_ * S_ * 2);
  u16* Ab = (u16*)alloc((size_t)B_ * S_ * NH_ * D_ * 2);

  auto cdiv = [](long a, long b) { return (int)((a + b - 1) / b); };

  // 1. downconvert
  k_f32_to_bf16<<<cdiv(nX, 256), 256, 0, stream>>>(hs, Xb, nX);
  k_f32_to_bf16<<<cdiv(nWq, 256), 256, 0, stream>>>(Wq, Wqb, nWq);
  k_f32_to_bf16<<<cdiv(nWkv, 256), 256, 0, stream>>>(Wk, Wkb, nWkv);
  k_f32_to_bf16<<<cdiv(nWkv, 256), 256, 0, stream>>>(Wv, Wvb, nWkv);
  k_f32_to_bf16<<<cdiv(nWq, 256), 256, 0, stream>>>(Wo, Wob, nWq);

  // 2. QKV projections: M = B*S = 4096
  dim3 blk(32, 8);
  dim3 gq(NH_ * D_ / 64, (B_ * S_) / 128);      // (32, 32)
  dim3 gkv(NKV_ * D_ / 64, (B_ * S_) / 128);    // (16, 32)
  k_gemm_bf16<<<gq, blk, 0, stream>>>(Xb, Wqb, Qf, B_ * S_, NH_ * D_, H_);
  k_gemm_bf16<<<gkv, blk, 0, stream>>>(Xb, Wkb, Kf, B_ * S_, NKV_ * D_, H_);
  k_gemm_bf16<<<gkv, blk, 0, stream>>>(Xb, Wvb, Vf, B_ * S_, NKV_ * D_, H_);

  // 3. RMS-norm + RoPE + transpose + quantize (fold d^-0.5 * log2e into Q)
  const float qscale = 0.1275174716f;           // log2(e) / sqrt(128)
  k_normrope<<<B_ * S_ * NH_, 32, 0, stream>>>(Qf, cosb, sinb, qnw, Qb, NH_, 1, 1, 0, qscale);
  k_normrope<<<B_ * S_ * NKV_, 32, 0, stream>>>(Kf, cosb, sinb, knw, Kb, NKV_, 1, 1, 0, 1.0f);
  k_normrope<<<B_ * S_ * NKV_, 32, 0, stream>>>(Vf, cosb, sinb, qnw, Vt, NKV_, 0, 0, 1, 1.0f);

  // 4. sliding-window flash attention
  dim3 ablk(32, 4);
  dim3 agrid(S_ / 64, B_ * NH_);                // (32, 32)
  k_attn<<<agrid, ablk, 0, stream>>>(Qb, Kb, Vt, Ab);

  // 5. output projection -> f32 d_out
  dim3 go(H_ / 64, (B_ * S_) / 128);            // (32, 32)
  k_gemm_bf16<<<go, blk, 0, stream>>>(Ab, Wob, out, B_ * S_, H_, H_);
}
